// get_model_40544491274530
// MI455X (gfx1250) — compile-verified, matrix operand order checked by reading, not run on hardware
//
#include <hip/hip_runtime.h>
#include <hip/hip_bf16.h>

typedef __attribute__((ext_vector_type(16))) _Float16 v16h;
typedef __attribute__((ext_vector_type(8)))  _Float16 v8h;
typedef __attribute__((ext_vector_type(8)))  float    v8f;

#define EPS_BN 1e-5f

// ---------------------------------------------------------------------------
// split xyz (B,N,7) into pos (B,N,3) and feat (B,N,4)
// ---------------------------------------------------------------------------
__global__ void split_kernel(const float* __restrict__ xyz,
                             float* __restrict__ pos,
                             float* __restrict__ feat, int BN) {
  int i = blockIdx.x * blockDim.x + threadIdx.x;
  if (i >= BN) return;
  const float* p = xyz + (size_t)i * 7;
  pos[(size_t)i * 3 + 0] = p[0];
  pos[(size_t)i * 3 + 1] = p[1];
  pos[(size_t)i * 3 + 2] = p[2];
  feat[(size_t)i * 4 + 0] = p[3];
  feat[(size_t)i * 4 + 1] = p[4];
  feat[(size_t)i * 4 + 2] = p[5];
  feat[(size_t)i * 4 + 3] = p[6];
}

// ---------------------------------------------------------------------------
// Farthest point sampling: one block per batch.
// Emits fps index (carry *before* update, matching lax.scan) and gathers
// new_xyz = pos[far].
// ---------------------------------------------------------------------------
__global__ void fps_kernel(const float* __restrict__ pos,
                           float* __restrict__ newpos,
                           int* __restrict__ fpsidx, int N, int S) {
  __shared__ float dist[4096];
  __shared__ float rmax[256];
  __shared__ int   rarg[256];
  __shared__ int   farS;
  int b = blockIdx.x;
  int t = threadIdx.x;
  int nt = blockDim.x;
  const float* P = pos + (size_t)b * N * 3;
  for (int i = t; i < N; i += nt) dist[i] = 1e10f;
  if (t == 0) farS = 0;
  __syncthreads();
  for (int it = 0; it < S; ++it) {
    int far = farS;
    if (t == 0) {
      fpsidx[(size_t)b * S + it] = far;
      newpos[((size_t)b * S + it) * 3 + 0] = P[(size_t)far * 3 + 0];
      newpos[((size_t)b * S + it) * 3 + 1] = P[(size_t)far * 3 + 1];
      newpos[((size_t)b * S + it) * 3 + 2] = P[(size_t)far * 3 + 2];
    }
    float cx = P[(size_t)far * 3 + 0];
    float cy = P[(size_t)far * 3 + 1];
    float cz = P[(size_t)far * 3 + 2];
    float bm = -1.0f; int ba = 0x7fffffff;
    for (int i = t; i < N; i += nt) {
      float dx = P[(size_t)i * 3 + 0] - cx;
      float dy = P[(size_t)i * 3 + 1] - cy;
      float dz = P[(size_t)i * 3 + 2] - cz;
      float d = dx * dx + dy * dy + dz * dz;
      float dd = fminf(dist[i], d);
      dist[i] = dd;
      if (dd > bm) { bm = dd; ba = i; }   // increasing i => first-max kept
    }
    rmax[t] = bm; rarg[t] = ba;
    __syncthreads();
    for (int o = nt >> 1; o > 0; o >>= 1) {
      if (t < o) {
        if (rmax[t + o] > rmax[t] ||
            (rmax[t + o] == rmax[t] && rarg[t + o] < rarg[t])) {
          rmax[t] = rmax[t + o]; rarg[t] = rarg[t + o];
        }
      }
      __syncthreads();
    }
    if (t == 0) farS = rarg[0];
    __syncthreads();
  }
}

// ---------------------------------------------------------------------------
// Ball query: one thread per (b,s) center; first K in-radius indices in
// ascending order, padded with the first hit (center itself always hits).
// ---------------------------------------------------------------------------
__global__ void ball_query_kernel(const float* __restrict__ pos,
                                  const float* __restrict__ newpos,
                                  int* __restrict__ out,
                                  int N, int S, int K, float r2, int BS) {
  int id = blockIdx.x * blockDim.x + threadIdx.x;
  if (id >= BS) return;
  int b = id / S;
  const float* P = pos + (size_t)b * N * 3;
  float cx = newpos[(size_t)id * 3 + 0];
  float cy = newpos[(size_t)id * 3 + 1];
  float cz = newpos[(size_t)id * 3 + 2];
  int* o = out + (size_t)id * K;
  int cnt = 0, first = -1;
  for (int n = 0; n < N && cnt < K; ++n) {
    float dx = P[(size_t)n * 3 + 0] - cx;
    float dy = P[(size_t)n * 3 + 1] - cy;
    float dz = P[(size_t)n * 3 + 2] - cz;
    float d = dx * dx + dy * dy + dz * dz;
    if (d <= r2) { if (first < 0) first = n; o[cnt++] = n; }
  }
  if (first < 0) first = 0;
  for (; cnt < K; ++cnt) o[cnt] = first;
}

// ---------------------------------------------------------------------------
// Gather + concat (points feats, rel xyz) -> f16 activation matrix for batch b
// X is (S*K) x Cpad row-major, zero-padded columns.
// ---------------------------------------------------------------------------
__global__ void group_kernel(const float* __restrict__ feat,
                             const float* __restrict__ pos,
                             const float* __restrict__ newpos,
                             const int* __restrict__ gi,
                             _Float16* __restrict__ X,
                             int b, int N, int S, int K, int Cp, int Cpad) {
  int i = blockIdx.x * blockDim.x + threadIdx.x;
  int tot = S * K * Cpad;
  if (i >= tot) return;
  int row = i / Cpad, c = i % Cpad;
  int s = row / K, k = row % K;
  int g = gi[((size_t)b * S + s) * K + k];
  float v = 0.0f;
  if (c < Cp) {
    v = feat[((size_t)b * N + g) * Cp + c];
  } else if (c < Cp + 3) {
    int cc = c - Cp;
    v = pos[((size_t)b * N + g) * 3 + cc] - newpos[((size_t)b * S + s) * 3 + cc];
  }
  X[(size_t)row * Cpad + c] = (_Float16)v;
}

// sa_group_all build: concat(xyz, points) over all Np points of batch b
__global__ void groupall_kernel(const float* __restrict__ pos,
                                const float* __restrict__ pts,
                                _Float16* __restrict__ X,
                                int b, int Np, int Cp, int Cpad) {
  int i = blockIdx.x * blockDim.x + threadIdx.x;
  int tot = Np * Cpad;
  if (i >= tot) return;
  int row = i / Cpad, c = i % Cpad;
  float v = 0.0f;
  if (c < 3)              v = pos[((size_t)b * Np + row) * 3 + c];
  else if (c < 3 + Cp)    v = pts[((size_t)b * Np + row) * Cp + (c - 3)];
  X[i] = (_Float16)v;
}

// f32 weight (D x C) -> f16 (Dpad x Cpad), zero-padded rows and cols
__global__ void wconv_kernel(const float* __restrict__ W,
                             _Float16* __restrict__ Wp,
                             int D, int C, int Cpad, int Dpad) {
  int i = blockIdx.x * blockDim.x + threadIdx.x;
  if (i >= Dpad * Cpad) return;
  int d = i / Cpad, c = i % Cpad;
  Wp[i] = (d < D && c < C) ? (_Float16)W[(size_t)d * C + c] : (_Float16)0.0f;
}

// ---------------------------------------------------------------------------
// Register-blocked WMMA GEMM + fused bias/BN/ReLU epilogue.
// One wave computes a 32(M) x 64(N) block of Y = X(M x Cpad) * Wp^T, using
// 2x4 = 8 f32 accumulators. Per 32-wide K step: 12 b128 loads feed 8 WMMAs
// (1.5 loads/WMMA vs 4 for an unblocked tile).
// A layout (16-bit 16x32): lane halves select K in {8h..8h+7, 16+8h..16+8h+7}.
// B column n == row n of Wp (same per-lane K pattern); Wp rows padded to 64.
// D layout: lane -> N=lane&15, vgpr r -> M = r + 8*(lane>>4).
// ---------------------------------------------------------------------------
__global__ void __launch_bounds__(32)
wmma_gemm_bn_relu(const _Float16* __restrict__ X,
                  const _Float16* __restrict__ Wp,
                  const float* __restrict__ bias,
                  const float* __restrict__ gamma,
                  const float* __restrict__ beta,
                  _Float16* __restrict__ Y,
                  int Cpad, int D, int Dp) {
  int lane = threadIdx.x & 31;
  int mt = blockIdx.x, nt = blockIdx.y;
  int half = lane >> 4, l16 = lane & 15;
  const _Float16* ar0 = X + (size_t)(mt * 32 + l16) * Cpad;
  const _Float16* ar1 = ar0 + (size_t)16 * Cpad;
  const _Float16* br0 = Wp + (size_t)(nt * 64 + l16) * Cpad;
  v8f acc[2][4];
#pragma unroll
  for (int mi = 0; mi < 2; ++mi)
#pragma unroll
    for (int ni = 0; ni < 4; ++ni)
      acc[mi][ni] = (v8f){0.f, 0.f, 0.f, 0.f, 0.f, 0.f, 0.f, 0.f};

  for (int k = 0; k < Cpad; k += 32) {
    union { v16h v; v8h h[2]; } a[2], bmat[4];
    a[0].h[0] = *(const v8h*)(ar0 + k + half * 8);
    a[0].h[1] = *(const v8h*)(ar0 + k + 16 + half * 8);
    a[1].h[0] = *(const v8h*)(ar1 + k + half * 8);
    a[1].h[1] = *(const v8h*)(ar1 + k + 16 + half * 8);
#pragma unroll
    for (int ni = 0; ni < 4; ++ni) {
      const _Float16* br = br0 + (size_t)(ni * 16) * Cpad;
      bmat[ni].h[0] = *(const v8h*)(br + k + half * 8);
      bmat[ni].h[1] = *(const v8h*)(br + k + 16 + half * 8);
    }
#pragma unroll
    for (int mi = 0; mi < 2; ++mi)
#pragma unroll
      for (int ni = 0; ni < 4; ++ni)
        acc[mi][ni] = __builtin_amdgcn_wmma_f32_16x16x32_f16(
            false, a[mi].v, false, bmat[ni].v, (short)0, acc[mi][ni], false,
            false);
  }

  const float inv = rsqrtf(1.0f + EPS_BN);
#pragma unroll
  for (int ni = 0; ni < 4; ++ni) {
    int n = nt * 64 + ni * 16 + l16;
    if (n >= Dp) continue;
    float bv = 0.f, sc = 0.f, bt = 0.f;
    if (n < D) {
      bv = bias[n];
      sc = gamma[n] * inv;
      bt = beta[n];
    }
#pragma unroll
    for (int mi = 0; mi < 2; ++mi) {
      _Float16* yr = Y + (size_t)(mt * 32 + mi * 16 + half * 8) * Dp + n;
#pragma unroll
      for (int r = 0; r < 8; ++r) {
        float y = (n < D) ? fmaxf((acc[mi][ni][r] + bv) * sc + bt, 0.0f) : 0.0f;
        yr[(size_t)r * Dp] = (_Float16)y;
      }
    }
  }
}

// max over K samples of (S*K x Dp) f16 -> out[b,s,chanOff+d] (B,S,Ctot) f32
__global__ void maxpool_kernel(const _Float16* __restrict__ Y,
                               float* __restrict__ out,
                               int b, int S, int K, int D, int Dp,
                               int chanOff, int Ctot) {
  int i = blockIdx.x * blockDim.x + threadIdx.x;
  if (i >= S * D) return;
  int s = i / D, d = i % D;
  const _Float16* p = Y + (size_t)(s * K) * Dp + d;
  float m = -1e30f;
  for (int k = 0; k < K; ++k) m = fmaxf(m, (float)p[(size_t)k * Dp]);
  out[((size_t)b * S + s) * Ctot + chanOff + d] = m;
}

// dense FC (f32): mode 1 = bias+BN+ReLU, mode 0 = bias only
__global__ void fc_kernel(const float* __restrict__ Xin,
                          const float* __restrict__ W,
                          const float* __restrict__ bias,
                          const float* __restrict__ gamma,
                          const float* __restrict__ beta,
                          float* __restrict__ Yout,
                          int R, int Cin, int Cout, int mode) {
  int i = blockIdx.x * blockDim.x + threadIdx.x;
  if (i >= R * Cout) return;
  int r = i / Cout, o = i % Cout;
  const float* x = Xin + (size_t)r * Cin;
  const float* w = W + (size_t)o * Cin;
  float acc = 0.0f;
  for (int c = 0; c < Cin; ++c) acc += x[c] * w[c];
  acc += bias[o];
  if (mode) {
    acc = acc * gamma[o] * rsqrtf(1.0f + EPS_BN) + beta[o];
    acc = fmaxf(acc, 0.0f);
  }
  Yout[i] = acc;
}

__global__ void copy_kernel(const float* __restrict__ src,
                            float* __restrict__ dst, int n) {
  int i = blockIdx.x * blockDim.x + threadIdx.x;
  if (i < n) dst[i] = src[i];
}

// ---------------------------------------------------------------------------
extern "C" void kernel_launch(void* const* d_in, const int* in_sizes, int n_in,
                              void* d_out, int out_size, void* d_ws, size_t ws_size,
                              hipStream_t stream) {
  (void)in_sizes; (void)n_in; (void)out_size; (void)ws_size;
  const int B = 8;
  char* ws = (char*)d_ws;
  size_t off = 0;
  auto alloc = [&](size_t bytes) -> char* {
    char* p = ws + off;
    off += (bytes + 255) & ~(size_t)255;
    return p;
  };

  float* pos1  = (float*)alloc((size_t)B * 4096 * 3 * 4);
  float* feat1 = (float*)alloc((size_t)B * 4096 * 4 * 4);
  float* posA  = (float*)alloc((size_t)B * 2048 * 3 * 4);
  float* posB  = (float*)alloc((size_t)B * 512 * 3 * 4);
  float* posC  = (float*)alloc((size_t)B * 128 * 3 * 4);
  float* l1p   = (float*)alloc((size_t)B * 2048 * 288 * 4);
  float* l2p   = (float*)alloc((size_t)B * 512 * 576 * 4);
  float* l3p   = (float*)alloc((size_t)B * 128 * 1152 * 4);
  float* l4    = (float*)alloc((size_t)B * 2048 * 4);
  float* ht0   = (float*)alloc((size_t)B * 2048 * 4);
  float* ht1   = (float*)alloc((size_t)B * 2048 * 4);
  int* fpsidx  = (int*)alloc((size_t)B * 2048 * 4);
  int* ballidx = (int*)alloc((size_t)B * 2048 * 48 * 4);   // max over stages
  _Float16* wp[3];
  wp[0] = (_Float16*)alloc((size_t)5 * 1024 * 1024);
  wp[1] = (_Float16*)alloc((size_t)5 * 1024 * 1024);
  wp[2] = (_Float16*)alloc((size_t)5 * 1024 * 1024);
  _Float16* bufA = (_Float16*)alloc((size_t)32 * 1024 * 1024);
  _Float16* bufB = (_Float16*)alloc((size_t)32 * 1024 * 1024);

  auto align32 = [](int x) { return (x + 31) & ~31; };
  auto align64 = [](int x) { return (x + 63) & ~63; };

  const float* xyz = (const float*)d_in[0];
  int BN = B * 4096;
  split_kernel<<<(BN + 255) / 256, 256, 0, stream>>>(xyz, pos1, feat1, BN);

  auto run_stage = [&](const float* posIn, const float* featIn, int N, int S,
                       int Cpts, float* posOut, float* ptsOut, int Ctot,
                       const float* radii, const int* Ks,
                       const int widths[3][3], int paramBase) {
    fps_kernel<<<B, 256, 0, stream>>>(posIn, posOut, fpsidx, N, S);
    int chanOff = 0;
    for (int br = 0; br < 3; ++br) {
      int K = Ks[br];
      float r2 = radii[br] * radii[br];
      int BS = B * S;
      ball_query_kernel<<<(BS + 255) / 256, 256, 0, stream>>>(
          posIn, posOut, ballidx, N, S, K, r2, BS);
      int Cin[4];
      Cin[0] = Cpts + 3;
      for (int l = 0; l < 3; ++l) Cin[l + 1] = widths[br][l];
      for (int l = 0; l < 3; ++l) {
        int D = widths[br][l], C = Cin[l], Cp = align32(C), Dpad = align64(D);
        const float* W = (const float*)d_in[paramBase + br * 12 + l * 4 + 0];
        int n = Dpad * Cp;
        wconv_kernel<<<(n + 255) / 256, 256, 0, stream>>>(W, wp[l], D, C, Cp,
                                                          Dpad);
      }
      int M = S * K;  // always a multiple of 32 here
      for (int b = 0; b < B; ++b) {
        {
          int Cp0 = align32(Cin[0]);
          int n = M * Cp0;
          group_kernel<<<(n + 255) / 256, 256, 0, stream>>>(
              featIn, posIn, posOut, ballidx, bufA, b, N, S, K, Cpts, Cp0);
        }
        _Float16* src = bufA;
        _Float16* dst = bufB;
        for (int l = 0; l < 3; ++l) {
          int D = widths[br][l], C = Cin[l], Cp = align32(C), Dp = align32(D);
          int Dpad = align64(D);
          const float* bias = (const float*)d_in[paramBase + br * 12 + l * 4 + 1];
          const float* gam  = (const float*)d_in[paramBase + br * 12 + l * 4 + 2];
          const float* bet  = (const float*)d_in[paramBase + br * 12 + l * 4 + 3];
          dim3 g((unsigned)(M / 32), (unsigned)(Dpad / 64));
          wmma_gemm_bn_relu<<<g, 32, 0, stream>>>(src, wp[l], bias, gam, bet,
                                                  dst, Cp, D, Dp);
          _Float16* t = src; src = dst; dst = t;
        }
        int D = widths[br][2], Dp = align32(D);
        int n = S * D;
        maxpool_kernel<<<(n + 255) / 256, 256, 0, stream>>>(
            src, ptsOut, b, S, K, D, Dp, chanOff, Ctot);
      }
      chanOff += widths[br][2];
    }
  };

  {
    const float r[3] = {0.05f, 0.1f, 0.2f};
    const int k[3] = {16, 32, 48};
    const int w[3][3] = {{32, 48, 64}, {48, 64, 96}, {64, 96, 128}};
    run_stage(pos1, feat1, 4096, 2048, 4, posA, l1p, 288, r, k, w, 1);
  }
  {
    const float r[3] = {0.1f, 0.2f, 0.4f};
    const int k[3] = {32, 48, 64};
    const int w[3][3] = {{64, 96, 128}, {96, 128, 192}, {128, 192, 256}};
    run_stage(posA, l1p, 2048, 512, 288, posB, l2p, 576, r, k, w, 37);
  }
  {
    const float r[3] = {0.2f, 0.4f, 0.6f};
    const int k[3] = {48, 64, 80};
    const int w[3][3] = {{128, 192, 256}, {192, 256, 384}, {256, 384, 512}};
    run_stage(posB, l2p, 512, 128, 576, posC, l3p, 1152, r, k, w, 73);
  }

  // ---- sa_group_all: concat(xyz, points) -> MLP [512,1024,2048] -> max ----
  {
    const int w4[3] = {512, 1024, 2048};
    int Cin4[4] = {1155, 512, 1024, 2048};
    for (int l = 0; l < 3; ++l) {
      int D = w4[l], C = Cin4[l], Cp = align32(C), Dpad = align64(D);
      const float* W = (const float*)d_in[109 + l * 4 + 0];
      int n = Dpad * Cp;
      wconv_kernel<<<(n + 255) / 256, 256, 0, stream>>>(W, wp[l], D, C, Cp,
                                                        Dpad);
    }
    int c4p = align32(1155);  // 1184
    for (int b = 0; b < B; ++b) {
      int n = 128 * c4p;
      groupall_kernel<<<(n + 255) / 256, 256, 0, stream>>>(posC, l3p, bufA, b,
                                                           128, 1152, c4p);
      _Float16* src = bufA;
      _Float16* dst = bufB;
      for (int l = 0; l < 3; ++l) {
        int D = w4[l], C = Cin4[l], Cp = align32(C), Dp = align32(D);
        int Dpad = align64(D);
        dim3 g((unsigned)(128 / 32), (unsigned)(Dpad / 64));
        wmma_gemm_bn_relu<<<g, 32, 0, stream>>>(
            src, wp[l],
            (const float*)d_in[109 + l * 4 + 1],
            (const float*)d_in[109 + l * 4 + 2],
            (const float*)d_in[109 + l * 4 + 3], dst, Cp, D, Dp);
        _Float16* t = src; src = dst; dst = t;
      }
      // src = 128 x 2048 (Dp==2048); max over 128 rows -> l4[b,:]
      maxpool_kernel<<<(2048 + 255) / 256, 256, 0, stream>>>(
          src, l4, b, 1, 128, 2048, 2048, 0, 2048);
    }
  }

  // ---- head ----
  {
    const int hd[4][2] = {{2048, 1024}, {1024, 768}, {768, 512}, {512, 256}};
    const float* hx = l4;
    float* hb[2] = {ht0, ht1};
    for (int i = 0; i < 4; ++i) {
      int Ci = hd[i][0], Co = hd[i][1];
      int n = B * Co;
      fc_kernel<<<(n + 255) / 256, 256, 0, stream>>>(
          hx, (const float*)d_in[121 + i * 4 + 0],
          (const float*)d_in[121 + i * 4 + 1],
          (const float*)d_in[121 + i * 4 + 2],
          (const float*)d_in[121 + i * 4 + 3], hb[i & 1], B, Ci, Co, 1);
      hx = hb[i & 1];
    }
    // fc5: 256 -> 3, logits into d_out[0..23]
    fc_kernel<<<1, 32, 0, stream>>>(hx, (const float*)d_in[137],
                                    (const float*)d_in[138], nullptr, nullptr,
                                    (float*)d_out, B, 256, 3, 0);
    // l4 into d_out[24..]
    copy_kernel<<<(B * 2048 + 255) / 256, 256, 0, stream>>>(
        l4, (float*)d_out + 24, B * 2048);
  }
}
